// HetGAT_critic_38474317038025
// MI455X (gfx1250) — matrix operand order repaired and checked by reading, not compiled, not run on hardware
//
#include <hip/hip_runtime.h>

// ---------------------------------------------------------------------------
// HetGAT critic for MI455X (gfx1250, wave32, WMMA bf16 16x16x32)
//
// B=4096, NA=32, DIU=128, DOU=32, DIB=256, DOB=64, DH=128, N=33, 3H=384
// All GEMMs via v_wmma_f32_16x16x32_bf16 (bf16 operands, f32 accumulate).
// GRU scan: one workgroup per 16 batch rows, weights resident in 320KB LDS,
// x_t staged with global_load_async_to_lds_b128 (double-buffered, ASYNCcnt).
// ---------------------------------------------------------------------------

#define B_    4096
#define NA_   32
#define DIU_  128
#define DOU_  32
#define DIB_  256
#define DOB_  64
#define DH_   128
#define G3_   384
#define NNODE 33

typedef __attribute__((ext_vector_type(16))) __bf16 bf16x16;
typedef __attribute__((ext_vector_type(8)))  float  f32x8;

union FragBF {
    bf16x16 v;
    uint4   q[2];
};

__device__ __forceinline__ unsigned short f2bf(float f) {
    unsigned int u = __float_as_uint(f);
    unsigned int r = (u + 0x7FFFu + ((u >> 16) & 1u)) >> 16;   // RNE
    return (unsigned short)r;
}
__device__ __forceinline__ float bf2f(unsigned short s) {
    return __uint_as_float(((unsigned int)s) << 16);
}
__device__ __forceinline__ float sigmoidf(float x) {
    return 1.0f / (1.0f + expf(-x));
}
__device__ __forceinline__ f32x8 v8zero() {
    f32x8 z;
#pragma unroll
    for (int i = 0; i < 8; ++i) z[i] = 0.0f;
    return z;
}

// gfx1250 async global->LDS copy (ASYNCcnt-tracked).  lds_off = LDS byte
// offset (flat addr low 32 bits), 16 bytes per lane.
__device__ __forceinline__ void asyncLoadLds128(unsigned lds_off,
                                                const void* gaddr) {
    asm volatile("global_load_async_to_lds_b128 %0, %1, off"
                 :: "v"(lds_off), "v"(gaddr) : "memory");
}
__device__ __forceinline__ void waitAsync0() {
    asm volatile("s_wait_asynccnt 0" ::: "memory");
}

// A fragment: 16x32 bf16 tile of row-major A (stride lda), tile rows m0..m0+15,
// K block k0..k0+31.  Lane L (r=L&15, half=L>>4): row m0+r,
// elems 0..7 = k0+half*8 .. +7, elems 8..15 = k0+half*8+16 .. +23.
__device__ __forceinline__ void loadFragA(FragBF& f, const unsigned short* A,
                                          int lda, int m0, int k0, int lane) {
    int r = lane & 15, half = lane >> 4;
    const unsigned short* p = A + (size_t)(m0 + r) * lda + k0 + half * 8;
    f.q[0] = *(const uint4*)(p);
    f.q[1] = *(const uint4*)(p + 16);
}

// B fragment: 32x16 bf16 tile from weights packed [N][K] (stride ldb=K).
// Lane L: col n0+r, elems 0..7 = k0+half*8 .. +7, elems 8..15 = k0+half*8+16 .. +23.
__device__ __forceinline__ void loadFragB(FragBF& f, const unsigned short* Bt,
                                          int ldb, int n0, int k0, int lane) {
    int r = lane & 15, half = lane >> 4;
    const unsigned short* p = Bt + (size_t)(n0 + r) * ldb + k0 + half * 8;
    f.q[0] = *(const uint4*)(p);
    f.q[1] = *(const uint4*)(p + 16);
}

// D/C layout: lane L holds col n0+(L&15); VGPR e holds row m0 + e + (L>=16?8:0)
__device__ __forceinline__ void storeTileF32(float* C, int ldc, int m0, int n0,
                                             int lane, const f32x8& c) {
    int r = lane & 15, half = lane >> 4;
    float* p = C + (size_t)(m0 + half * 8) * ldc + n0 + r;
#pragma unroll
    for (int e = 0; e < 8; ++e) p[(size_t)e * ldc] = c[e];
}

__device__ __forceinline__ f32x8 wmma_bf16(const FragBF& a, const FragBF& b,
                                           const f32x8& c) {
    return __builtin_amdgcn_wmma_f32_16x16x32_bf16(
        false, a.v, false, b.v, (short)0, c, false, false);
}

// ---------------------------------------------------------------------------
// Generic WMMA GEMM: C[M,N] = act( A[M,K](bf16,row-major) * Bt[N,K]^T + bias )
// One wave computes a 16x64 tile (4 accumulators) -> each A fragment is
// reused by 4 WMMAs.  N must be a multiple of 64.  ACT: 0 none,1 relu,2 sigm.
// ---------------------------------------------------------------------------
template <int ACT, int OUTBF>
__global__ void __launch_bounds__(128) gemm_bf16_kernel(
    const unsigned short* __restrict__ A, int lda,
    const unsigned short* __restrict__ Bt,
    const float* __restrict__ bias,
    void* __restrict__ Cout, int M, int N, int K) {
    int wid     = blockIdx.x * (blockDim.x >> 5) + (threadIdx.x >> 5);
    int lane    = threadIdx.x & 31;
    int ngroups = N >> 6;
    int mtiles  = M >> 4;
    if (wid >= mtiles * ngroups) return;      // uniform per-wave
    int mt = wid / ngroups, ng = wid % ngroups;
    int m0 = mt << 4, n0g = ng << 6;

    f32x8 acc[4];
#pragma unroll
    for (int j = 0; j < 4; ++j) acc[j] = v8zero();

    int kSteps = K >> 5;
    for (int ks = 0; ks < kSteps; ++ks) {
        FragBF a;
        loadFragA(a, A, lda, m0, ks * 32, lane);
#pragma unroll
        for (int j = 0; j < 4; ++j) {
            FragBF b;
            loadFragB(b, Bt, K, n0g + j * 16, ks * 32, lane);
            acc[j] = wmma_bf16(a, b, acc[j]);
        }
    }

    int r = lane & 15, half = lane >> 4;
#pragma unroll
    for (int j = 0; j < 4; ++j) {
        int n = n0g + j * 16 + r;
        float bn = bias ? bias[n] : 0.0f;
#pragma unroll
        for (int e = 0; e < 8; ++e) {
            float x = acc[j][e] + bn;
            if (ACT == 1) x = fmaxf(x, 0.0f);
            if (ACT == 2) x = sigmoidf(x);
            acc[j][e] = x;
        }
        if (OUTBF) {
            unsigned short* C = (unsigned short*)Cout;
            unsigned short* p = C + (size_t)(m0 + half * 8) * N + n;
#pragma unroll
            for (int e = 0; e < 8; ++e) p[(size_t)e * N] = f2bf(acc[j][e]);
        } else {
            float* C = (float*)Cout;
            float* p = C + (size_t)(m0 + half * 8) * N + n;
#pragma unroll
            for (int e = 0; e < 8; ++e) p[(size_t)e * N] = acc[j][e];
        }
    }
}

// ---------------------------------------------------------------------------
// GRU scan (UE path): X bf16 [B,NA,DH] -> Y bf16 [B,NA,DH].
// One block = 16 batch rows, 256 threads (8 waves). Weights live in LDS
// (2 x 384x128 bf16 = 192KB on CDNA5's 320KB WGP LDS).  x_t is staged with
// async global->LDS copies, double-buffered so the t+1 stage overlaps the
// WMMA + gate math of step t.
// ---------------------------------------------------------------------------
__global__ void __launch_bounds__(256, 1) gru_scan_kernel(
    const unsigned short* __restrict__ X,
    const float* __restrict__ wih, const float* __restrict__ whh,
    const float* __restrict__ bih, const float* __restrict__ bhh,
    unsigned short* __restrict__ Y) {
    __shared__ __attribute__((aligned(16))) unsigned short sWih[G3_ * DH_];
    __shared__ __attribute__((aligned(16))) unsigned short sWhh[G3_ * DH_];
    __shared__ __attribute__((aligned(16))) unsigned short sX[2][16 * DH_];
    __shared__ __attribute__((aligned(16))) unsigned short sH[16 * DH_];
    __shared__ __attribute__((aligned(16))) float sGi[16 * G3_];
    __shared__ __attribute__((aligned(16))) float sGh[16 * G3_];
    __shared__ float sBih[G3_];
    __shared__ float sBhh[G3_];

    const int tid  = threadIdx.x;
    const int lane = tid & 31;
    const int w    = tid >> 5;
    const int bm0  = blockIdx.x * 16;

    for (int i = tid; i < G3_ * DH_; i += 256) {
        sWih[i] = f2bf(wih[i]);
        sWhh[i] = f2bf(whh[i]);
    }
    for (int i = tid; i < G3_; i += 256) { sBih[i] = bih[i]; sBhh[i] = bhh[i]; }
    for (int i = tid; i < 16 * DH_; i += 256) sH[i] = 0;

    // this thread's 16-byte slice of the x-tile stage
    const int xi  = tid >> 4;
    const int xc0 = (tid & 15) * 8;

    // kick off async stage of x_0 into buffer 0
    {
        const unsigned short* g = X + ((size_t)(bm0 + xi) * NA_ + 0) * DH_ + xc0;
        unsigned lo = (unsigned)(size_t)&sX[0][xi * DH_ + xc0];
        asyncLoadLds128(lo, g);
    }
    __syncthreads();

    for (int t = 0; t < NA_; ++t) {
        const int buf = t & 1;
        waitAsync0();           // my slice of x_t landed in LDS
        __syncthreads();        // everyone's slice landed

        // overlap: stage x_{t+1} into the other buffer while we compute
        if (t + 1 < NA_) {
            const unsigned short* g =
                X + ((size_t)(bm0 + xi) * NA_ + (t + 1)) * DH_ + xc0;
            unsigned lo = (unsigned)(size_t)&sX[buf ^ 1][xi * DH_ + xc0];
            asyncLoadLds128(lo, g);
        }

        // Gi / Gh via WMMA: 8 waves x 3 N-tiles cover 384 gate columns
        f32x8 accI[3], accH[3];
#pragma unroll
        for (int j = 0; j < 3; ++j) { accI[j] = v8zero(); accH[j] = v8zero(); }
#pragma unroll
        for (int ks = 0; ks < 4; ++ks) {
            FragBF aX, aH;
            loadFragA(aX, sX[buf], DH_, 0, ks * 32, lane);
            loadFragA(aH, sH, DH_, 0, ks * 32, lane);
#pragma unroll
            for (int j = 0; j < 3; ++j) {
                int n0 = (w * 3 + j) * 16;
                FragBF bI, bH;
                loadFragB(bI, sWih, DH_, n0, ks * 32, lane);
                loadFragB(bH, sWhh, DH_, n0, ks * 32, lane);
                accI[j] = wmma_bf16(aX, bI, accI[j]);
                accH[j] = wmma_bf16(aH, bH, accH[j]);
            }
        }
#pragma unroll
        for (int j = 0; j < 3; ++j) {
            int n0 = (w * 3 + j) * 16;
            storeTileF32(sGi, G3_, 0, n0, lane, accI[j]);
            storeTileF32(sGh, G3_, 0, n0, lane, accH[j]);
        }
        __syncthreads();

        // gate math: 2048 h elements, 8 per thread
#pragma unroll
        for (int e = 0; e < 8; ++e) {
            int idx = (tid << 3) + e;
            int m = idx >> 7, c = idx & 127;
            float gir = sGi[m * G3_ + c]       + sBih[c];
            float ghr = sGh[m * G3_ + c]       + sBhh[c];
            float giz = sGi[m * G3_ + 128 + c] + sBih[128 + c];
            float ghz = sGh[m * G3_ + 128 + c] + sBhh[128 + c];
            float gin = sGi[m * G3_ + 256 + c] + sBih[256 + c];
            float ghn = sGh[m * G3_ + 256 + c] + sBhh[256 + c];
            float r = sigmoidf(gir + ghr);
            float z = sigmoidf(giz + ghz);
            float n = tanhf(gin + r * ghn);
            float hnew = (1.0f - z) * n + z * bf2f(sH[m * DH_ + c]);
            unsigned short hb = f2bf(hnew);
            sH[m * DH_ + c] = hb;
            Y[((size_t)(bm0 + m) * NA_ + t) * DH_ + c] = hb;
        }
        __syncthreads();
    }
}

// ---------------------------------------------------------------------------
// BS path: single GRU step from h=0:  h = (1-z)*n, gh = bhh.
// gi (f32, already has bih from GEMM bias) -> h bf16.
// ---------------------------------------------------------------------------
__global__ void gru_zero_step_kernel(const float* __restrict__ gi,
                                     const float* __restrict__ bhh,
                                     unsigned short* __restrict__ H, int M) {
    int idx = blockIdx.x * blockDim.x + threadIdx.x;
    if (idx >= M * DH_) return;
    int m = idx >> 7, c = idx & 127;
    const float* g = gi + (size_t)m * G3_;
    float r = sigmoidf(g[c] + bhh[c]);
    float z = sigmoidf(g[128 + c] + bhh[128 + c]);
    float n = tanhf(g[256 + c] + r * bhh[256 + c]);
    H[idx] = f2bf((1.0f - z) * n);
}

// ---------------------------------------------------------------------------
// Packing / conversion kernels
// ---------------------------------------------------------------------------
__global__ void pack_xcat_ue_kernel(const float* __restrict__ s,
                                    const float* __restrict__ a,
                                    unsigned short* __restrict__ out) {
    size_t idx = (size_t)blockIdx.x * blockDim.x + threadIdx.x;
    const size_t total = (size_t)B_ * NA_ * (DIU_ + DOU_);
    if (idx >= total) return;
    int col = (int)(idx % (DIU_ + DOU_));
    size_t row = idx / (DIU_ + DOU_);
    int b = (int)(row >> 5), nI = (int)(row & 31);
    float v = (col < DIU_)
                  ? s[(size_t)b * (DIU_ * NA_ + DIB_) + nI * DIU_ + col]
                  : a[(size_t)b * (DOU_ * NA_ + DOB_) + nI * DOU_ + (col - DIU_)];
    out[idx] = f2bf(v);
}

__global__ void pack_xcat_bs_kernel(const float* __restrict__ s,
                                    const float* __restrict__ a,
                                    unsigned short* __restrict__ out) {
    size_t idx = (size_t)blockIdx.x * blockDim.x + threadIdx.x;
    const size_t total = (size_t)B_ * (DIB_ + DOB_);
    if (idx >= total) return;
    int col = (int)(idx % (DIB_ + DOB_));
    int b   = (int)(idx / (DIB_ + DOB_));
    float v = (col < DIB_)
                  ? s[(size_t)b * (DIU_ * NA_ + DIB_) + DIU_ * NA_ + col]
                  : a[(size_t)b * (DOU_ * NA_ + DOB_) + DOU_ * NA_ + (col - DIB_)];
    out[idx] = f2bf(v);
}

// W f32 [K][N] -> Wt bf16 [N][K]
__global__ void tconv_kernel(const float* __restrict__ W,
                             unsigned short* __restrict__ Wt, int K, int N) {
    int idx = blockIdx.x * blockDim.x + threadIdx.x;
    if (idx >= N * K) return;
    int n = idx / K, k = idx % K;
    Wt[idx] = f2bf(W[(size_t)k * N + n]);
}

__global__ void conv_kernel(const float* __restrict__ W,
                            unsigned short* __restrict__ O, int n) {
    int idx = blockIdx.x * blockDim.x + threadIdx.x;
    if (idx < n) O[idx] = f2bf(W[idx]);
}

__global__ void pack_emb_kernel(const unsigned short* __restrict__ embUe,
                                const unsigned short* __restrict__ embBs,
                                unsigned short* __restrict__ out) {
    size_t idx = (size_t)blockIdx.x * blockDim.x + threadIdx.x;
    const size_t total = (size_t)B_ * NNODE * DH_;
    if (idx >= total) return;
    int c = (int)(idx & 127);
    size_t row = idx >> 7;
    int b = (int)(row / NNODE), i = (int)(row % NNODE);
    out[idx] = (i < NA_) ? embUe[((size_t)b * NA_ + i) * DH_ + c]
                         : embBs[(size_t)b * DH_ + c];
}

// ---------------------------------------------------------------------------
// GAT attention + ELU + eval head, one block (128 thr) per batch element.
// att = softmax over dim=1; einsum('bij,bjf->bf') == colsum(att) . Wh
// ---------------------------------------------------------------------------
__global__ void __launch_bounds__(128) gat_final_kernel(
    const float* __restrict__ Wh, const float* __restrict__ gat_a,
    const float* __restrict__ adj, const float* __restrict__ ev1w,
    const float* __restrict__ ev1b, const float* __restrict__ ev2w,
    const float* __restrict__ ev2b, float* __restrict__ out) {
    __shared__ float sWh[NNODE * DH_];
    __shared__ float sA[2 * DH_];
    __shared__ float sW1[NNODE], sW2[NNODE], sCol[NNODE];
    __shared__ float sHp[DH_], sXv[DH_], sRed[DH_];

    int b = blockIdx.x, tid = threadIdx.x;
    for (int i = tid; i < NNODE * DH_; i += 128)
        sWh[i] = Wh[(size_t)b * NNODE * DH_ + i];
    for (int i = tid; i < 2 * DH_; i += 128) sA[i] = gat_a[i];
    __syncthreads();

    if (tid < NNODE) {
        float w1 = 0.0f, w2 = 0.0f;
        for (int c = 0; c < DH_; ++c) {
            float v = sWh[tid * DH_ + c];
            w1 += v * sA[c];
            w2 += v * sA[DH_ + c];
        }
        sW1[tid] = w1;
        sW2[tid] = w2;
    }
    __syncthreads();

    if (tid < NNODE) {                      // column-j softmax over i
        int j = tid;
        float mx = -3.0e38f;
        for (int i = 0; i < NNODE; ++i) {
            float e = sW1[i] + sW2[j];
            e = (e > 0.0f) ? e : 0.01f * e;             // leaky_relu
            float me = (adj[i * NNODE + j] > 0.0f) ? e : -9.0e15f;
            mx = fmaxf(mx, me);
        }
        float ssum = 0.0f;
        for (int i = 0; i < NNODE; ++i) {
            float e = sW1[i] + sW2[j];
            e = (e > 0.0f) ? e : 0.01f * e;
            float me = (adj[i * NNODE + j] > 0.0f) ? e : -9.0e15f;
            ssum += expf(me - mx);
        }
        float cs = 0.0f;
        for (int i = 0; i < NNODE; ++i) {
            float e = sW1[i] + sW2[j];
            e = (e > 0.0f) ? e : 0.01f * e;
            float me = (adj[i * NNODE + j] > 0.0f) ? e : -9.0e15f;
            cs += expf(me - mx) / ssum;
        }
        sCol[j] = cs;
    }
    __syncthreads();

    {   // h_prime[f] = sum_j colsum[j]*Wh[j,f]; ELU
        float hp = 0.0f;
        for (int j = 0; j < NNODE; ++j) hp += sCol[j] * sWh[j * DH_ + tid];
        hp = (hp > 0.0f) ? hp : (expf(hp) - 1.0f);
        sHp[tid] = hp;
    }
    __syncthreads();

    {   // x = relu(hp @ ev1_w + ev1_b)
        float x = ev1b[tid];
        for (int k = 0; k < DH_; ++k) x += sHp[k] * ev1w[k * DH_ + tid];
        sXv[tid] = fmaxf(x, 0.0f);
    }
    __syncthreads();

    sRed[tid] = sXv[tid] * ev2w[tid];
    __syncthreads();
    for (int s = 64; s > 0; s >>= 1) {
        if (tid < s) sRed[tid] += sRed[tid + s];
        __syncthreads();
    }
    if (tid == 0) out[b] = sRed[0] + ev2b[0];
}

// ---------------------------------------------------------------------------
// Launch
// ---------------------------------------------------------------------------
extern "C" void kernel_launch(void* const* d_in, const int* in_sizes, int n_in,
                              void* d_out, int out_size, void* d_ws,
                              size_t ws_size, hipStream_t stream) {
    (void)in_sizes; (void)n_in; (void)out_size; (void)ws_size;
    const float* s_in    = (const float*)d_in[0];
    const float* a_in    = (const float*)d_in[1];
    const float* ue_l1_w = (const float*)d_in[2];
    const float* ue_l1_b = (const float*)d_in[3];
    const float* ue_l2_w = (const float*)d_in[4];
    const float* ue_l2_b = (const float*)d_in[5];
    const float* ue_wih0 = (const float*)d_in[6];
    const float* ue_whh0 = (const float*)d_in[7];
    const float* ue_bih0 = (const float*)d_in[8];
    const float* ue_bhh0 = (const float*)d_in[9];
    const float* ue_wih1 = (const float*)d_in[10];
    const float* ue_whh1 = (const float*)d_in[11];
    const float* ue_bih1 = (const float*)d_in[12];
    const float* ue_bhh1 = (const float*)d_in[13];
    const float* bs_l1_w = (const float*)d_in[14];
    const float* bs_l1_b = (const float*)d_in[15];
    const float* bs_l2_w = (const float*)d_in[16];
    const float* bs_l2_b = (const float*)d_in[17];
    const float* bs_wih0 = (const float*)d_in[18];
    const float* bs_bih0 = (const float*)d_in[20];
    const float* bs_bhh0 = (const float*)d_in[21];
    const float* bs_wih1 = (const float*)d_in[22];
    const float* bs_bih1 = (const float*)d_in[24];
    const float* bs_bhh1 = (const float*)d_in[25];
    const float* gat_w   = (const float*)d_in[26];
    const float* gat_a   = (const float*)d_in[27];
    const float* adj     = (const float*)d_in[28];
    const float* ev1_w   = (const float*)d_in[29];
    const float* ev1_b   = (const float*)d_in[30];
    const float* ev2_w   = (const float*)d_in[31];
    const float* ev2_b   = (const float*)d_in[32];
    float* out = (float*)d_out;

    const size_t MB = 1ull << 20;
    char* ws = (char*)d_ws;
    unsigned short* bufA = (unsigned short*)(ws);            // 72MB ping
    unsigned short* bufB = (unsigned short*)(ws + 72 * MB);  // 36MB pong
    char* C = ws + 108 * MB;
    unsigned short* XcatBs = (unsigned short*)(C);
    unsigned short* H1bs   = (unsigned short*)(C + 3 * MB);
    unsigned short* X0bs   = (unsigned short*)(C + 4 * MB);
    unsigned short* H0bs   = (unsigned short*)(C + 5 * MB);
    unsigned short* EMBbs  = (unsigned short*)(C + 6 * MB);
    float*          GIbs   = (float*)(C + 7 * MB);
    char* WP = C + 14 * MB;
    unsigned short* W_UE_L1T  = (unsigned short*)(WP);
    unsigned short* W_UE_L2T  = (unsigned short*)(WP + 512 * 1024);
    unsigned short* W_BS_L1T  = (unsigned short*)(WP + 1 * MB);
    unsigned short* W_BS_L2T  = (unsigned short*)(WP + 1536 * 1024);
    unsigned short* W_GATT    = (unsigned short*)(WP + 2 * MB);
    unsigned short* W_BS_WIH0 = (unsigned short*)(WP + 2560 * 1024);
    unsigned short* W_BS_WIH1 = (unsigned short*)(WP + 3 * MB);

    // --- weight packing ------------------------------------------------
    tconv_kernel<<<(128 * 160 + 255) / 256, 256, 0, stream>>>(ue_l1_w, W_UE_L1T, 160, 128);
    tconv_kernel<<<(128 * 128 + 255) / 256, 256, 0, stream>>>(ue_l2_w, W_UE_L2T, 128, 128);
    tconv_kernel<<<(128 * 320 + 255) / 256, 256, 0, stream>>>(bs_l1_w, W_BS_L1T, 320, 128);
    tconv_kernel<<<(128 * 128 + 255) / 256, 256, 0, stream>>>(bs_l2_w, W_BS_L2T, 128, 128);
    tconv_kernel<<<(128 * 128 + 255) / 256, 256, 0, stream>>>(gat_w,   W_GATT,   128, 128);
    conv_kernel<<<(G3_ * DH_ + 255) / 256, 256, 0, stream>>>(bs_wih0, W_BS_WIH0, G3_ * DH_);
    conv_kernel<<<(G3_ * DH_ + 255) / 256, 256, 0, stream>>>(bs_wih1, W_BS_WIH1, G3_ * DH_);

    // --- input packing -------------------------------------------------
    {
        size_t tot = (size_t)B_ * NA_ * 160;
        pack_xcat_ue_kernel<<<(unsigned)((tot + 255) / 256), 256, 0, stream>>>(s_in, a_in, bufA);
    }
    {
        size_t tot = (size_t)B_ * 320;
        pack_xcat_bs_kernel<<<(unsigned)((tot + 255) / 256), 256, 0, stream>>>(s_in, a_in, XcatBs);
    }

    const int MU = B_ * NA_;  // 131072 rows in UE path
    auto gblocks = [](int M, int N) { return ((M >> 4) * (N >> 6) + 3) / 4; };

    // --- UE path: l1 (relu) -> l2 (sigmoid) ----------------------------
    gemm_bf16_kernel<1, 1><<<gblocks(MU, 128), 128, 0, stream>>>(
        bufA, 160, W_UE_L1T, ue_l1_b, bufB, MU, 128, 160);
    gemm_bf16_kernel<2, 1><<<gblocks(MU, 128), 128, 0, stream>>>(
        bufB, 128, W_UE_L2T, ue_l2_b, bufA, MU, 128, 128);

    // --- UE path: 2-layer GRU scan over NA=32 --------------------------
    gru_scan_kernel<<<B_ / 16, 256, 0, stream>>>(bufA, ue_wih0, ue_whh0,
                                                 ue_bih0, ue_bhh0, bufB);
    gru_scan_kernel<<<B_ / 16, 256, 0, stream>>>(bufB, ue_wih1, ue_whh1,
                                                 ue_bih1, ue_bhh1, bufA);

    // --- BS path -------------------------------------------------------
    gemm_bf16_kernel<1, 1><<<gblocks(B_, 128), 128, 0, stream>>>(
        XcatBs, 320, W_BS_L1T, bs_l1_b, H1bs, B_, 128, 320);
    gemm_bf16_kernel<2, 1><<<gblocks(B_, 128), 128, 0, stream>>>(
        H1bs, 128, W_BS_L2T, bs_l2_b, X0bs, B_, 128, 128);
    gemm_bf16_kernel<0, 0><<<gblocks(B_, G3_), 128, 0, stream>>>(
        X0bs, 128, W_BS_WIH0, bs_bih0, GIbs, B_, G3_, 128);
    gru_zero_step_kernel<<<(B_ * DH_ + 255) / 256, 256, 0, stream>>>(
        GIbs, bs_bhh0, H0bs, B_);
    gemm_bf16_kernel<0, 0><<<gblocks(B_, G3_), 128, 0, stream>>>(
        H0bs, 128, W_BS_WIH1, bs_bih1, GIbs, B_, G3_, 128);
    gru_zero_step_kernel<<<(B_ * DH_ + 255) / 256, 256, 0, stream>>>(
        GIbs, bs_bhh1, EMBbs, B_);

    // --- concat embeddings, GAT projection -----------------------------
    {
        size_t tot = (size_t)B_ * NNODE * DH_;
        pack_emb_kernel<<<(unsigned)((tot + 255) / 256), 256, 0, stream>>>(bufA, EMBbs, bufB);
    }
    const int MG = B_ * NNODE;  // 135168
    float* Wh = (float*)bufA;   // 69.2MB fits in 72MB region
    gemm_bf16_kernel<0, 0><<<gblocks(MG, 128), 128, 0, stream>>>(
        bufB, 128, W_GATT, nullptr, Wh, MG, 128, 128);

    // --- attention + ELU + eval head -----------------------------------
    gat_final_kernel<<<B_, 128, 0, stream>>>(Wh, gat_a, adj, ev1_w, ev1_b,
                                             ev2_w, ev2_b, out);
}
